// Attention_25434796327401
// MI455X (gfx1250) — compile-verified, matrix operand order checked by reading, not run on hardware
//
#include <hip/hip_runtime.h>
#include <hip/hip_bf16.h>
#include <cstdint>

typedef __attribute__((ext_vector_type(16))) __bf16 v16bf;
typedef __attribute__((ext_vector_type(8)))  __bf16 v8bf;
typedef __attribute__((ext_vector_type(8)))  float  v8f;

#define HH     16
#define WW     64
#define NTOK   1024          // HH*WW
#define EMB    768
#define HEADS  12
#define HD     64
#define BATCH  32
#define WK_R   5             // 11//2 (w radius); h radius = 3

// ---------------------------------------------------------------------------
// Transpose + f32 -> bf16 convert:  Wt[c][k] = W[k][c]
// ---------------------------------------------------------------------------
__global__ void transpose_to_bf16(const float* __restrict__ W,
                                  __bf16* __restrict__ Wt,
                                  int rows /*K*/, int cols /*N*/) {
  int idx = blockIdx.x * blockDim.x + threadIdx.x;
  if (idx < rows * cols) {
    int c = idx % cols;
    int k = idx / cols;
    Wt[(size_t)c * rows + k] = (__bf16)W[idx];
  }
}

// ---------------------------------------------------------------------------
// QKV GEMM. Block = 4 waves sharing one 32-row A slab staged in LDS (f32 ->
// bf16 once). Each wave computes 32x64: 8 accumulators, each A frag feeds 4
// WMMAs and each B frag feeds 2 WMMAs (1 global b128 load per WMMA).
// Scatters Q (scaled), K, V^T as bf16.
// ---------------------------------------------------------------------------
__global__ __launch_bounds__(128)
void qkv_gemm(const float* __restrict__ x, const __bf16* __restrict__ Wt,
              __bf16* __restrict__ Qb, __bf16* __restrict__ Kb,
              __bf16* __restrict__ Vt) {
  __shared__ __align__(16) __bf16 sA[32 * EMB];    // 48 KB

  const int tid  = threadIdx.x;
  const int lane = tid & 31;
  const int wv   = tid >> 5;
  const int row0 = blockIdx.x * 32;

  // stage A slab: 32*768 contiguous f32 -> bf16 in LDS
  const float4* src = (const float4*)(x + (size_t)row0 * EMB);
  for (int i = tid; i < 32 * EMB / 4; i += 128) {
    float4 f = src[i];
    __bf16* d = sA + i * 4;
    d[0] = (__bf16)f.x; d[1] = (__bf16)f.y;
    d[2] = (__bf16)f.z; d[3] = (__bf16)f.w;
  }
  __syncthreads();

  const int col0 = (blockIdx.y * 4 + wv) * 64;     // one head segment
  const int r  = lane & 15;
  const int kb = (lane >> 4) << 3;
  const __bf16* arow0 = sA + r * EMB;
  const __bf16* arow1 = sA + (16 + r) * EMB;

  v8f acc[2][4] = {};
  for (int k0 = 0; k0 < EMB; k0 += 32) {
    v16bf a0, a1;
    {
      v8bf lo = *(const v8bf*)(arow0 + k0 + kb);
      v8bf hi = *(const v8bf*)(arow0 + k0 + kb + 16);
#pragma unroll
      for (int i = 0; i < 8; ++i) { a0[i] = lo[i]; a0[i + 8] = hi[i]; }
    }
    {
      v8bf lo = *(const v8bf*)(arow1 + k0 + kb);
      v8bf hi = *(const v8bf*)(arow1 + k0 + kb + 16);
#pragma unroll
      for (int i = 0; i < 8; ++i) { a1[i] = lo[i]; a1[i + 8] = hi[i]; }
    }
#pragma unroll
    for (int t = 0; t < 4; ++t) {
      const __bf16* brow = Wt + (size_t)(col0 + t * 16 + r) * EMB;
      v16bf bfr;
      v8bf blo = *(const v8bf*)(brow + k0 + kb);
      v8bf bhi = *(const v8bf*)(brow + k0 + kb + 16);
#pragma unroll
      for (int i = 0; i < 8; ++i) { bfr[i] = blo[i]; bfr[i + 8] = bhi[i]; }
      acc[0][t] = __builtin_amdgcn_wmma_f32_16x16x32_bf16(
          false, a0, false, bfr, (short)0, acc[0][t], false, false);
      acc[1][t] = __builtin_amdgcn_wmma_f32_16x16x32_bf16(
          false, a1, false, bfr, (short)0, acc[1][t], false, false);
    }
  }

  const int which = col0 / EMB;                    // 0=q 1=k 2=v (uniform)
  const int head  = (col0 % EMB) / HD;             // uniform
  const int ncol  = lane & 15;
  const int mhalf = (lane >> 4) << 3;
#pragma unroll
  for (int mtt = 0; mtt < 2; ++mtt) {
#pragma unroll
    for (int t = 0; t < 4; ++t) {
      const int dbase = t * 16;
#pragma unroll
      for (int j = 0; j < 8; ++j) {
        const int tokrow = row0 + mtt * 16 + j + mhalf;
        const int bidx = tokrow >> 10;
        const int tok  = tokrow & 1023;
        const size_t bh = (size_t)(bidx * HEADS + head);
        const float v = acc[mtt][t][j];
        if (which == 0) {
          Qb[(bh * NTOK + tok) * HD + dbase + ncol] = (__bf16)(v * 0.125f);
        } else if (which == 1) {
          Kb[(bh * NTOK + tok) * HD + dbase + ncol] = (__bf16)v;
        } else {
          Vt[(bh * HD + dbase + ncol) * NTOK + tok] = (__bf16)v;
        }
      }
    }
  }
}

// ---------------------------------------------------------------------------
// Local-window attention, flash-style online softmax.
// Block = (b, head, image-row h); 4 waves = the 4 query tiles of that row,
// sharing a K window (<= 7 rows x 64 tokens x 64d bf16) staged in LDS via
// async global->LDS copies.
// ---------------------------------------------------------------------------
__global__ __launch_bounds__(128)
void attn_kernel(const __bf16* __restrict__ Qb, const __bf16* __restrict__ Kb,
                 const __bf16* __restrict__ Vt, __bf16* __restrict__ AO) {
  __shared__ __align__(16) __bf16 sK[7 * WW * HD];   // 57344 B
  __shared__ __align__(16) __bf16 sP[4][16 * 32];    //  4096 B

  const int tid  = threadIdx.x;
  const int lane = tid & 31;
  const int wv   = tid >> 5;                 // query w-tile 0..3
  const int bid  = blockIdx.x;
  const int h    = bid & 15;
  const int head = (bid >> 4) % HEADS;
  const int b    = bid / (16 * HEADS);

  const int h0 = (h - 3 > 0) ? h - 3 : 0;
  const int h1 = (h + 3 < HH - 1) ? h + 3 : HH - 1;
  const int nh = h1 - h0 + 1;

  const size_t bh = (size_t)(b * HEADS + head);
  const __bf16* kwin  = Kb + (bh * NTOK + (size_t)h0 * WW) * HD;
  const __bf16* vbase = Vt + bh * HD * NTOK;

  // ---- async stage of the K window (contiguous nh*64 tokens) ----
  {
    const unsigned sKbase = (unsigned)(uintptr_t)(void*)sK;
    const unsigned long long gbase = (unsigned long long)(uintptr_t)kwin;
    const int nchunks = nh * WW * HD * 2 / 16;        // 16B chunks
    for (int c = tid; c < nchunks; c += 128) {
      unsigned dst = sKbase + (unsigned)c * 16u;
      unsigned long long srca = gbase + (unsigned long long)c * 16ull;
      asm volatile("global_load_async_to_lds_b128 %0, %1, off"
                   :: "v"(dst), "v"(srca) : "memory");
    }
    asm volatile("s_wait_asynccnt 0x0" ::: "memory");
  }
  __syncthreads();

  // ---- per-wave setup ----
  const int wbase = wv * 16;
  const int qt = h * 4 + wv;
  const int wt0 = ((wbase - WK_R) > 0) ? ((wbase - WK_R) >> 4) : 0;
  const int wt1m = (wbase + 15 + WK_R) >> 4;
  const int wt1 = (wt1m < 3) ? wt1m : 3;
  const int nw  = wt1 - wt0 + 1;
  const int nkt = nh * nw;                   // <= 21 key tiles
  const int npair = (nkt + 1) >> 1;

  const int r     = lane & 15;
  const int kb    = (lane >> 4) << 3;
  const int ncol  = lane & 15;
  const int mhalf = (lane >> 4) << 3;

  // Q A-operands (K=64 -> two 32-chunks), pre-scaled by d^-1/2 in Qb
  const __bf16* qbase = Qb + (bh * NTOK + (size_t)qt * 16) * HD;
  v16bf qa[2];
#pragma unroll
  for (int c = 0; c < 2; ++c) {
    const __bf16* row = qbase + r * HD + c * 32;
    v8bf lo = *(const v8bf*)(row + kb);
    v8bf hi = *(const v8bf*)(row + kb + 16);
#pragma unroll
    for (int i = 0; i < 8; ++i) { qa[c][i] = lo[i]; qa[c][i + 8] = hi[i]; }
  }

  float mrun[8], lrun[8];
#pragma unroll
  for (int j = 0; j < 8; ++j) { mrun[j] = -1e30f; lrun[j] = 0.f; }
  v8f o[4] = {};
  __bf16* myP = &sP[wv][0];

  for (int p = 0; p < npair; ++p) {
    const int t0 = 2 * p, t1 = 2 * p + 1;
    const bool valid1 = (t1 < nkt);          // uniform per wave

    // ---- score tiles from LDS-staged K ----
    const int hp0 = h0 + t0 / nw, wta = wt0 + t0 % nw;
    const int kl0 = (hp0 - h0) * WW + wta * 16;
    v8f s0 = {};
#pragma unroll
    for (int c = 0; c < 2; ++c) {
      const __bf16* row = sK + (size_t)(kl0 + r) * HD + c * 32;
      v16bf bk;
      v8bf lo = *(const v8bf*)(row + kb);
      v8bf hi = *(const v8bf*)(row + kb + 16);
#pragma unroll
      for (int i = 0; i < 8; ++i) { bk[i] = lo[i]; bk[i + 8] = hi[i]; }
      s0 = __builtin_amdgcn_wmma_f32_16x16x32_bf16(false, qa[c], false, bk,
                                                   (short)0, s0, false, false);
    }
    int hp1 = hp0, wtb = wta;
    v8f s1;
    if (valid1) {
      hp1 = h0 + t1 / nw; wtb = wt0 + t1 % nw;
      const int kl1 = (hp1 - h0) * WW + wtb * 16;
      v8f z = {};
#pragma unroll
      for (int c = 0; c < 2; ++c) {
        const __bf16* row = sK + (size_t)(kl1 + r) * HD + c * 32;
        v16bf bk;
        v8bf lo = *(const v8bf*)(row + kb);
        v8bf hi = *(const v8bf*)(row + kb + 16);
#pragma unroll
        for (int i = 0; i < 8; ++i) { bk[i] = lo[i]; bk[i + 8] = hi[i]; }
        z = __builtin_amdgcn_wmma_f32_16x16x32_bf16(false, qa[c], false, bk,
                                                    (short)0, z, false, false);
      }
      s1 = z;
    } else {
#pragma unroll
      for (int j = 0; j < 8; ++j) s1[j] = -1e30f;
    }

    // ---- mask + online softmax update (row reductions across 16 lanes) ----
    const int wk0 = wta * 16 + ncol;
    const int wk1 = wtb * 16 + ncol;
    float scl[8];
#pragma unroll
    for (int j = 0; j < 8; ++j) {
      const int wq = wbase + j + mhalf;
      float a0 = s0[j];
      { const int d0 = wq - wk0; if (d0 > WK_R || d0 < -WK_R) a0 = -1e30f; }
      float a1 = s1[j];
      if (valid1) { const int d1 = wq - wk1;
                    if (d1 > WK_R || d1 < -WK_R) a1 = -1e30f; }
      float pm = fmaxf(a0, a1);
#pragma unroll
      for (int st = 1; st < 16; st <<= 1) pm = fmaxf(pm, __shfl_xor(pm, st, 32));
      const float mnew = fmaxf(mrun[j], pm);
      scl[j] = __expf(mrun[j] - mnew);
      const float p0 = __expf(a0 - mnew);
      const float p1 = __expf(a1 - mnew);
      float ps = p0 + p1;
#pragma unroll
      for (int st = 1; st < 16; st <<= 1) ps += __shfl_xor(ps, st, 32);
      lrun[j] = lrun[j] * scl[j] + ps;
      mrun[j] = mnew;
      myP[(j + mhalf) * 32 + ncol]      = (__bf16)p0;
      myP[(j + mhalf) * 32 + 16 + ncol] = (__bf16)p1;
    }
#pragma unroll
    for (int dt = 0; dt < 4; ++dt)
#pragma unroll
      for (int j = 0; j < 8; ++j) o[dt][j] *= scl[j];

    asm volatile("s_wait_dscnt 0x0" ::: "memory");   // wave-local LDS RAW

    // ---- P (A-operand from LDS) @ V (B-operand from global, transposed) ----
    v16bf pa;
    {
      const __bf16* row = myP + r * 32;
      v8bf lo = *(const v8bf*)(row + kb);
      v8bf hi = *(const v8bf*)(row + kb + 16);
#pragma unroll
      for (int i = 0; i < 8; ++i) { pa[i] = lo[i]; pa[i + 8] = hi[i]; }
    }
    const int k0g = hp0 * WW + wta * 16;       // global token bases
    const int k1g = valid1 ? hp1 * WW + wtb * 16 : k0g;
#pragma unroll
    for (int dt = 0; dt < 4; ++dt) {
      const __bf16* vrow = vbase + (size_t)(dt * 16 + ncol) * NTOK;
      v16bf bv;
      v8bf lo = *(const v8bf*)(vrow + k0g + kb);
      v8bf hi = *(const v8bf*)(vrow + k1g + kb);
#pragma unroll
      for (int i = 0; i < 8; ++i) { bv[i] = lo[i]; bv[i + 8] = hi[i]; }
      o[dt] = __builtin_amdgcn_wmma_f32_16x16x32_bf16(false, pa, false, bv,
                                                      (short)0, o[dt],
                                                      false, false);
    }
    asm volatile("s_wait_dscnt 0x0" ::: "memory");   // before sP reuse (WAR)
  }

  // ---- normalize + write [B*N, EMB] bf16 ----
#pragma unroll
  for (int dt = 0; dt < 4; ++dt) {
#pragma unroll
    for (int j = 0; j < 8; ++j) {
      const int tok = qt * 16 + j + mhalf;
      AO[((size_t)(b * NTOK + tok)) * EMB + head * HD + dt * 16 + ncol] =
          (__bf16)(o[dt][j] / lrun[j]);
    }
  }
}

// ---------------------------------------------------------------------------
// Output projection. A slab (32x768 bf16) staged via async global->LDS
// copies, shared by 4 waves; each wave computes 32x64 + bias -> f32 out.
// ---------------------------------------------------------------------------
__global__ __launch_bounds__(128)
void proj_gemm(const __bf16* __restrict__ A, const __bf16* __restrict__ Wt,
               const float* __restrict__ bias, float* __restrict__ out) {
  __shared__ __align__(16) __bf16 sA[32 * EMB];    // 48 KB

  const int tid  = threadIdx.x;
  const int lane = tid & 31;
  const int wv   = tid >> 5;
  const int row0 = blockIdx.x * 32;

  {
    const unsigned sAbase = (unsigned)(uintptr_t)(void*)sA;
    const unsigned long long gbase =
        (unsigned long long)(uintptr_t)(A + (size_t)row0 * EMB);
    for (int i = tid; i < 32 * EMB * 2 / 16; i += 128) {   // 3072 chunks
      unsigned dst = sAbase + (unsigned)i * 16u;
      unsigned long long srca = gbase + (unsigned long long)i * 16ull;
      asm volatile("global_load_async_to_lds_b128 %0, %1, off"
                   :: "v"(dst), "v"(srca) : "memory");
    }
    asm volatile("s_wait_asynccnt 0x0" ::: "memory");
  }
  __syncthreads();

  const int col0 = (blockIdx.y * 4 + wv) * 64;
  const int r  = lane & 15;
  const int kb = (lane >> 4) << 3;
  const __bf16* arow0 = sA + r * EMB;
  const __bf16* arow1 = sA + (16 + r) * EMB;

  v8f acc[2][4] = {};
  for (int k0 = 0; k0 < EMB; k0 += 32) {
    v16bf a0, a1;
    {
      v8bf lo = *(const v8bf*)(arow0 + k0 + kb);
      v8bf hi = *(const v8bf*)(arow0 + k0 + kb + 16);
#pragma unroll
      for (int i = 0; i < 8; ++i) { a0[i] = lo[i]; a0[i + 8] = hi[i]; }
    }
    {
      v8bf lo = *(const v8bf*)(arow1 + k0 + kb);
      v8bf hi = *(const v8bf*)(arow1 + k0 + kb + 16);
#pragma unroll
      for (int i = 0; i < 8; ++i) { a1[i] = lo[i]; a1[i + 8] = hi[i]; }
    }
#pragma unroll
    for (int t = 0; t < 4; ++t) {
      const __bf16* brow = Wt + (size_t)(col0 + t * 16 + r) * EMB;
      v16bf bfr;
      v8bf blo = *(const v8bf*)(brow + k0 + kb);
      v8bf bhi = *(const v8bf*)(brow + k0 + kb + 16);
#pragma unroll
      for (int i = 0; i < 8; ++i) { bfr[i] = blo[i]; bfr[i + 8] = bhi[i]; }
      acc[0][t] = __builtin_amdgcn_wmma_f32_16x16x32_bf16(
          false, a0, false, bfr, (short)0, acc[0][t], false, false);
      acc[1][t] = __builtin_amdgcn_wmma_f32_16x16x32_bf16(
          false, a1, false, bfr, (short)0, acc[1][t], false, false);
    }
  }

  const int ncol = lane & 15, mhalf = (lane >> 4) << 3;
#pragma unroll
  for (int mtt = 0; mtt < 2; ++mtt) {
#pragma unroll
    for (int t = 0; t < 4; ++t) {
      const float bb = bias[col0 + t * 16 + ncol];
#pragma unroll
      for (int j = 0; j < 8; ++j) {
        out[(size_t)(row0 + mtt * 16 + j + mhalf) * EMB + col0 + t * 16 + ncol] =
            acc[mtt][t][j] + bb;
      }
    }
  }
}

// ---------------------------------------------------------------------------
extern "C" void kernel_launch(void* const* d_in, const int* in_sizes, int n_in,
                              void* d_out, int out_size, void* d_ws,
                              size_t ws_size, hipStream_t stream) {
  const float* x      = (const float*)d_in[0];   // [32,1024,768]
  const float* W_qkv  = (const float*)d_in[1];   // [768,2304]
  const float* W_proj = (const float*)d_in[2];   // [768,768]
  const float* b_proj = (const float*)d_in[3];   // [768]
  float* out = (float*)d_out;

  size_t off = 0;
  auto take = [&](size_t bytes) -> void* {
    void* p = (char*)d_ws + off;
    off += (bytes + 255) & ~(size_t)255;
    return p;
  };
  __bf16* Wtq = (__bf16*)take((size_t)2304 * EMB * 2);
  __bf16* Wtp = (__bf16*)take((size_t)EMB * EMB * 2);
  __bf16* Qb  = (__bf16*)take((size_t)BATCH * HEADS * NTOK * HD * 2);
  __bf16* Kb  = (__bf16*)take((size_t)BATCH * HEADS * NTOK * HD * 2);
  __bf16* Vt  = (__bf16*)take((size_t)BATCH * HEADS * HD * NTOK * 2);
  __bf16* AO  = (__bf16*)take((size_t)BATCH * NTOK * EMB * 2);

  {
    int n1 = EMB * 3 * EMB;
    transpose_to_bf16<<<(n1 + 255) / 256, 256, 0, stream>>>(W_qkv, Wtq, EMB,
                                                            3 * EMB);
    int n2 = EMB * EMB;
    transpose_to_bf16<<<(n2 + 255) / 256, 256, 0, stream>>>(W_proj, Wtp, EMB,
                                                            EMB);
  }
  qkv_gemm<<<dim3(1024, 9), 128, 0, stream>>>(x, Wtq, Qb, Kb, Vt);
  attn_kernel<<<BATCH * HEADS * HH, 128, 0, stream>>>(Qb, Kb, Vt, AO);
  proj_gemm<<<dim3(1024, 3), 128, 0, stream>>>(AO, Wtp, b_proj, out);
}
